// PrimitiveDecoder_49675591746069
// MI455X (gfx1250) — matrix operand order repaired and checked by reading, not compiled, not run on hardware
//
#include <hip/hip_runtime.h>
#include <hip/hip_bf16.h>
#include <stdint.h>

// ---------------------------------------------------------------------------
// MI455X (gfx1250) persistent LSTM decoder.
//   - weights converted once to bf16, packed into WMMA A-operand tile layout
//   - per-layer weight slice staged to LDS via global_load_async_to_lds_b128
//   - v_wmma_f32_16x16x32_bf16 GEMV (column 0 of D used), fp32 accumulate
//   - one persistent kernel, 404 device-wide barriers (L2 atomics + s_sleep)
// ---------------------------------------------------------------------------

#define NZ        1024
#define NH        1024
#define NSTATE    64
#define NLAYERS   4
#define T_STEPS   101
#define NWG       128
#define NTHREADS  256

typedef __attribute__((ext_vector_type(16))) __bf16 v16bf;
typedef __attribute__((ext_vector_type(8)))  float  v8f;

// workspace layout (bytes)
#define WS_BAR   0                              // 2 x uint32 barrier state
#define WS_BIAS  256                            // 4*4096 f32 = 65536 B
#define WS_GBUF  (256 + 65536)                  // 2*4096 f32 (parity dbl-buf)
#define WS_WPK   (256 + 65536 + 32768)          // 65536 tiles * 1KB = 64 MB bf16

// ---------------------------------------------------------------------------
__global__ void k_init(const float* __restrict__ bih, const float* __restrict__ bhh,
                       float* __restrict__ bias, unsigned* __restrict__ bar) {
  int i = blockIdx.x * blockDim.x + threadIdx.x;
  if (i < 2) bar[i] = 0u;
  if (i < NLAYERS * 4 * NH) bias[i] = bih[i] + bhh[i];
}

// Pack fp32 weights -> bf16 WMMA A tiles.
// tile index i = wg*512 + layer*128 + rt*64 + mat*32 + chunk   (65536 tiles)
// Within a tile (16 rows x 32 K, bf16): element e: lane = e/16, t = e%16,
//   row-in-tile m = lane&15,  k = (lane/16)*16 + t
// so each lane's 16 bf16 are contiguous (two b128 loads in the GEMV).
__global__ void k_pack(const float* __restrict__ Wih, const float* __restrict__ Whh,
                       __bf16* __restrict__ wpk) {
  unsigned i = blockIdx.x;
  int chunk = i & 31;
  int mat   = (i >> 5) & 1;
  int rt    = (i >> 6) & 1;
  int layer = (i >> 7) & 3;
  int wg    = i >> 9;
  int gate  = wg >> 5, rb = wg & 31;
  const float* src = (mat ? Whh : Wih) + (size_t)layer * 4096 * 1024;
  __bf16* dst = wpk + (size_t)i * 512;
  for (int e = threadIdx.x; e < 512; e += blockDim.x) {
    int lane = e >> 4, tt = e & 15;
    int m = lane & 15;
    int k = ((lane >> 4) << 4) | tt;
    int row = gate * 1024 + rb * 32 + rt * 16 + m;
    int col = chunk * 32 + k;
    dst[e] = (__bf16)src[(size_t)row * 1024 + col];
  }
}

// ---------------------------------------------------------------------------
__device__ __forceinline__ float sigmoidf_(float x) { return 1.0f / (1.0f + __expf(-x)); }

__device__ __forceinline__ void grid_sync(unsigned* bar) {
  __syncthreads();
  if (threadIdx.x == 0) {
    unsigned g = __hip_atomic_load(&bar[1], __ATOMIC_RELAXED, __HIP_MEMORY_SCOPE_AGENT);
    unsigned a = __hip_atomic_fetch_add(&bar[0], 1u, __ATOMIC_ACQ_REL, __HIP_MEMORY_SCOPE_AGENT);
    if (a == NWG - 1) {
      __hip_atomic_store(&bar[0], 0u, __ATOMIC_RELAXED, __HIP_MEMORY_SCOPE_AGENT);
      __hip_atomic_fetch_add(&bar[1], 1u, __ATOMIC_RELEASE, __HIP_MEMORY_SCOPE_AGENT);
    } else {
      while (__hip_atomic_load(&bar[1], __ATOMIC_ACQUIRE, __HIP_MEMORY_SCOPE_AGENT) == g)
        __builtin_amdgcn_s_sleep(2);
    }
  }
  __syncthreads();
}

// Stage this WG's 128KB per-layer weight slice into LDS (async DMA path).
__device__ __forceinline__ void issue_slice_async(const __bf16* gsrc, uint32_t lds_base) {
  for (int i = threadIdx.x; i < 8192; i += NTHREADS) {   // 8192 x 16B = 128 KB
    uint32_t la = lds_base + (uint32_t)i * 16u;
    uint64_t ga = (uint64_t)(uintptr_t)gsrc + (uint64_t)i * 16u;
    asm volatile("global_load_async_to_lds_b128 %0, %1, off"
                 :: "v"(la), "v"(ga) : "memory");
  }
}

__global__ void __launch_bounds__(NTHREADS)
k_lstm(const float* __restrict__ z,  const float* __restrict__ Ws,
       const float* __restrict__ bs, const float* __restrict__ Wp,
       const float* __restrict__ bp, const float* __restrict__ bias,
       const __bf16* __restrict__ wpk, float* __restrict__ gbuf,
       unsigned* __restrict__ bar, float* __restrict__ out)
{
  extern __shared__ __align__(128) char smem[];
  __bf16* ldsA = (__bf16*)smem;                         // 128 KB weight slice
  __shared__ __align__(32) float  cL[NLAYERS][NH];      // full cell state (redundant per WG)
  __shared__ __align__(32) __bf16 hL[NLAYERS][NH];      // full hidden state, bf16
  __shared__ __align__(32) __bf16 zb[NZ];
  __shared__ float part[8][16];
  __shared__ float red[NTHREADS];

  const int tid  = threadIdx.x;
  const int wg   = blockIdx.x;
  const int gate = wg >> 5, rb = wg & 31;               // this WG: one gate, rows rb*32..+31
  const int lane = tid & 31;
  // scalarize all wave-uniform control values -> SGPR branches, EXEC untouched
  const int wave = __builtin_amdgcn_readfirstlane(tid >> 5);
  const int rt   = wave >> 2;    // which 16-row tile of our 32 rows
  const int q    = wave & 3;     // which quarter of the 64 combined K-chunks
  const int matw = q >> 1;       // 0 = Wih (x path), 1 = Whh (h path) -- wave constant
  const int cb   = (q & 1) * 16; // first chunk of this wave's 16-chunk run
  // low 32 bits of the flat shared address == LDS byte offset (aperture in high bits)
  const uint32_t ldsA_base = (uint32_t)(uintptr_t)(void*)ldsA;
  // per-wave A base: tiles [rt*64 + matw*32 + cb ...], each tile 512 bf16
  const __bf16* aBase = ldsA + (size_t)(rt * 64 + matw * 32 + cb) * 512 + lane * 16;
  const int bLane = ((lane >> 4) << 4);                 // lanes 0-15: K lo half, 16-31: hi half

  for (int e = tid; e < NH; e += NTHREADS) {
    zb[e] = (__bf16)z[e];
    for (int l = 0; l < NLAYERS; ++l) { cL[l][e] = 0.0f; hL[l][e] = (__bf16)0.0f; }
  }
  __syncthreads();

  // prologue: stage layer-0 slice
  issue_slice_async(wpk + (size_t)(wg * 4 + 0) * 128 * 512, ldsA_base);

  for (int t = 0; t < T_STEPS; ++t) {
    for (int l = 0; l < NLAYERS; ++l) {
      asm volatile("s_wait_asynccnt 0" ::: "memory");   // slice DMA done
      __syncthreads();

      v8f acc = {0.f, 0.f, 0.f, 0.f, 0.f, 0.f, 0.f, 0.f};
      // layer-0 input is z at t=0 and the zero vector afterwards
      const bool active = !(matw == 0 && l == 0 && t > 0);   // fully scalar condition
      if (active) {
        const __bf16* xsrc = (matw == 0) ? ((l == 0) ? zb : hL[l - 1]) : hL[l];
        const __bf16* bBase = xsrc + cb * 32 + bLane;
#pragma unroll
        for (int cc = 0; cc < 16; ++cc) {
          v16bf a = *(const v16bf*)(aBase + cc * 512);  // ds b128 x2, const offsets
          v16bf b = *(const v16bf*)(bBase + cc * 32);   // ds b128 x2, const offsets
          acc = __builtin_amdgcn_wmma_f32_16x16x32_bf16(false, a, false, b,
                                                        (short)0, acc, false, false);
        }
      }
      // column 0 of D: lane 0 holds rows 0..7, lane 16 holds rows 8..15
      if (lane == 0) {
        for (int r = 0; r < 8; ++r) part[wave][r] = acc[r];
      } else if (lane == 16) {
        for (int r = 0; r < 8; ++r) part[wave][8 + r] = acc[r];
      }
      __syncthreads();                                  // all LDS A reads complete

      // overlap: stage next layer's slice while we barrier + do elementwise
      issue_slice_async(wpk + (size_t)(wg * 4 + ((l + 1) & 3)) * 128 * 512, ldsA_base);

      if (tid < 32) {                                   // combine 4 K-quarters + bias
        int rtt = tid >> 4, m = tid & 15;
        float s = 0.0f;
        for (int qq = 0; qq < 4; ++qq) s += part[rtt * 4 + qq][m];
        int row = gate * 1024 + rb * 32 + tid;
        s += bias[l * 4096 + row];
        gbuf[(l & 1) * 4096 + row] = s;                 // publish pre-activation
      }
      __threadfence();
      grid_sync(bar);

      // full elementwise LSTM update, redundant per WG -> h stays local in LDS
      const float* gb = gbuf + (l & 1) * 4096;
      for (int e = tid; e < NH; e += NTHREADS) {
        float gi = gb[e], gf = gb[1024 + e], gg = gb[2048 + e], go = gb[3072 + e];
        float cn = sigmoidf_(gf) * cL[l][e] + sigmoidf_(gi) * tanhf(gg);
        float hn = sigmoidf_(go) * tanhf(cn);
        cL[l][e] = cn;
        hL[l][e] = (__bf16)hn;
      }
      __syncthreads();
    }

    // ---- heads (overlap with next step of other WGs; use local hL[3]) ----
    if (wg >= 32 && wg < 96) {                          // states row = wg-32
      int row = wg - 32;
      float s = 0.0f;
      for (int cix = tid; cix < NH; cix += NTHREADS)
        s += Ws[(size_t)row * NH + cix] * (float)hL[3][cix];
      red[tid] = s; __syncthreads();
      for (int st = NTHREADS >> 1; st > 0; st >>= 1) {
        if (tid < st) red[tid] += red[tid + st];
        __syncthreads();
      }
      if (tid == 0) out[(size_t)t * NSTATE + row] = red[0] + bs[row];
      __syncthreads();
    } else if (wg == 96) {                              // probs + samples
      int rowp = tid >> 7;                              // 0 or 1
      float s = 0.0f;
      for (int cix = (tid & 127); cix < NH; cix += 128)
        s += Wp[(size_t)rowp * NH + cix] * (float)hL[3][cix];
      red[tid] = s; __syncthreads();
      for (int st = 64; st > 0; st >>= 1) {
        if ((tid & 127) < st) red[tid] += red[tid + st];
        __syncthreads();
      }
      if (tid == 0) {
        float l0 = red[0]   + bp[0] + 1.0f;             // P_BIAS on logit 0
        float l1 = red[128] + bp[1];
        float mx = fmaxf(l0, l1);
        float e0 = __expf(l0 - mx), e1 = __expf(l1 - mx);
        float inv = 1.0f / (e0 + e1);
        float p0 = e0 * inv, p1 = e1 * inv;
        if (t == T_STEPS - 1) { p0 = 0.0f; p1 = 1.0f; }
        float* probs = out + (size_t)T_STEPS * NSTATE;
        probs[2 * t]     = p0;
        probs[2 * t + 1] = p1;
        long long* samp = (long long*)(out + (size_t)T_STEPS * NSTATE + (size_t)T_STEPS * 2);
        samp[t] = (t == T_STEPS - 1) ? 1 : 0;
      }
      __syncthreads();
    }
  }
}

// ---------------------------------------------------------------------------
extern "C" void kernel_launch(void* const* d_in, const int* in_sizes, int n_in,
                              void* d_out, int out_size, void* d_ws, size_t ws_size,
                              hipStream_t stream) {
  (void)in_sizes; (void)n_in; (void)out_size; (void)ws_size;
  const float* z   = (const float*)d_in[0];
  const float* Wih = (const float*)d_in[1];
  const float* Whh = (const float*)d_in[2];
  const float* bih = (const float*)d_in[3];
  const float* bhh = (const float*)d_in[4];
  const float* Ws  = (const float*)d_in[5];
  const float* bs  = (const float*)d_in[6];
  const float* Wp  = (const float*)d_in[7];
  const float* bp  = (const float*)d_in[8];

  char* ws = (char*)d_ws;
  unsigned* bar = (unsigned*)(ws + WS_BAR);
  float*    bias = (float*)(ws + WS_BIAS);
  float*    gbuf = (float*)(ws + WS_GBUF);
  __bf16*   wpk  = (__bf16*)(ws + WS_WPK);   // needs ~64 MB of workspace
  float*    outf = (float*)d_out;

  k_init<<<64, 256, 0, stream>>>(bih, bhh, bias, bar);
  k_pack<<<65536, 256, 0, stream>>>(Wih, Whh, wpk);

  (void)hipFuncSetAttribute((const void*)k_lstm,
                            hipFuncAttributeMaxDynamicSharedMemorySize, 131072);
  k_lstm<<<NWG, NTHREADS, 131072, stream>>>(z, Ws, bs, Wp, bp, bias, wpk, gbuf, bar, outf);
}
